// PerceiverBackbone_4767413699221
// MI455X (gfx1250) — matrix operand order repaired and checked by reading, not compiled, not run on hardware
//
#include <hip/hip_runtime.h>
#include <hip/hip_bf16.h>
#include <math.h>

// ---------------- model constants ----------------
#define BB   8
#define TT   4096
#define DIN  256
#define DM   512
#define LQN  256
#define HN   8
#define HDN  64
#define NBK  6
#define HIDN 2048

typedef __attribute__((ext_vector_type(16))) __bf16 v16bf;
typedef __attribute__((ext_vector_type(8)))  __bf16 v8bf;
typedef __attribute__((ext_vector_type(8)))  float  v8f;

__device__ __forceinline__ v16bf ld16(const __bf16* lo, const __bf16* hi) {
  v8bf a = *reinterpret_cast<const v8bf*>(lo);
  v8bf b = *reinterpret_cast<const v8bf*>(hi);
  v16bf r;
#pragma unroll
  for (int i = 0; i < 8; ++i) { r[i] = a[i]; r[i + 8] = b[i]; }
  return r;
}

__device__ __forceinline__ v8f wmma_bf16(v16bf a, v16bf b, v8f c) {
  return __builtin_amdgcn_wmma_f32_16x16x32_bf16(false, a, false, b, (short)0, c,
                                                 false, false);
}

__device__ __forceinline__ float gelu_exact(float x) {
  return 0.5f * x * (1.0f + erff(x * 0.70710678118654752f));
}

// ---------------- GEMM: C = act(A[M,K] @ W[N,K]^T + bias) (+ residual) ------
// block 256 threads (8 waves); block tile 256(M) x 64(N); wave tile 32x64
// (8 accumulators, 2 A-frags x 4 B-frags -> 8 WMMA / K-step); K step 32;
// double-buffered LDS so next-chunk global loads overlap the WMMA section.
template <int ACT, int RES>
__launch_bounds__(256)
__global__ void gemm_bias_kernel(const float* __restrict__ A,
                                 const float* __restrict__ W,
                                 const float* __restrict__ bias,
                                 const float* __restrict__ Res,
                                 float* __restrict__ C, int M, int N, int K) {
  __shared__ __align__(16) __bf16 As[2][256][40];
  __shared__ __align__(16) __bf16 Bs[2][64][40];
  const int tid = threadIdx.x;
  const int w = tid >> 5;
  const int l = tid & 31;
  const int hi = l >> 4;
  const int ln16 = l & 15;
  const int m0 = blockIdx.y * 256;
  const int n0 = blockIdx.x * 64;

  float4 ra[8], rb[2];

  auto issue_loads = [&](int kk) {
#pragma unroll
    for (int j = 0; j < 8; ++j) {        // A tile 256x32 -> 2048 float4 slots
      int idx = tid + 256 * j;
      ra[j] = *reinterpret_cast<const float4*>(
          &A[(size_t)(m0 + (idx >> 3)) * K + kk + ((idx & 7) << 2)]);
    }
#pragma unroll
    for (int j = 0; j < 2; ++j) {        // W tile 64x32 -> 512 float4 slots
      int idx = tid + 256 * j;
      rb[j] = *reinterpret_cast<const float4*>(
          &W[(size_t)(n0 + (idx >> 3)) * K + kk + ((idx & 7) << 2)]);
    }
  };

  auto store_tile = [&](int p) {
#pragma unroll
    for (int j = 0; j < 8; ++j) {
      int idx = tid + 256 * j;
      __bf16* d = &As[p][idx >> 3][(idx & 7) << 2];
      d[0] = (__bf16)ra[j].x; d[1] = (__bf16)ra[j].y;
      d[2] = (__bf16)ra[j].z; d[3] = (__bf16)ra[j].w;
    }
#pragma unroll
    for (int j = 0; j < 2; ++j) {
      int idx = tid + 256 * j;
      __bf16* d = &Bs[p][idx >> 3][(idx & 7) << 2];
      d[0] = (__bf16)rb[j].x; d[1] = (__bf16)rb[j].y;
      d[2] = (__bf16)rb[j].z; d[3] = (__bf16)rb[j].w;
    }
  };

  v8f acc[8];
#pragma unroll
  for (int a = 0; a < 8; ++a)
#pragma unroll
    for (int r = 0; r < 8; ++r) acc[a][r] = 0.0f;

  issue_loads(0);
  store_tile(0);
  __syncthreads();

  int p = 0;
  for (int kk = 0; kk < K; kk += 32) {
    const bool more = (kk + 32) < K;
    if (more) issue_loads(kk + 32);      // overlaps with the WMMAs below
    if ((kk + 64) < K) {                 // gfx1250 global_prefetch_b8
      __builtin_prefetch(&A[(size_t)(m0 + (tid >> 1)) * K + kk + 64 +
                            ((tid & 1) << 4)], 0, 1);
    }

    const __bf16* a0r = &As[p][w * 32 + ln16][0];
    const __bf16* a1r = &As[p][w * 32 + 16 + ln16][0];
    v16bf a0 = ld16(a0r + 8 * hi, a0r + 8 * hi + 16);
    v16bf a1 = ld16(a1r + 8 * hi, a1r + 8 * hi + 16);
#pragma unroll
    for (int nt = 0; nt < 4; ++nt) {
      const __bf16* br = &Bs[p][nt * 16 + ln16][0];
      v16bf bf = ld16(br + 8 * hi, br + 8 * hi + 16);
      acc[nt]     = wmma_bf16(a0, bf, acc[nt]);
      acc[4 + nt] = wmma_bf16(a1, bf, acc[4 + nt]);
    }

    if (more) store_tile(p ^ 1);         // waits on the batched loads here
    __syncthreads();
    p ^= 1;
  }

  // epilogue: C-layout M = r + 8*hi, N = ln16
#pragma unroll
  for (int half = 0; half < 2; ++half)
#pragma unroll
    for (int nt = 0; nt < 4; ++nt) {
      const int col = n0 + nt * 16 + ln16;
      const float bv = bias[col];
#pragma unroll
      for (int r = 0; r < 8; ++r) {
        const int row = m0 + w * 32 + half * 16 + r + (hi << 3);
        float v = acc[half * 4 + nt][r] + bv;
        if (ACT == 1) v = gelu_exact(v);
        if (RES) v += Res[(size_t)row * N + col];
        C[(size_t)row * N + col] = v;
      }
    }
}

// ---------------- flash attention -------------------------------------------
// grid: (B*H, LQ/128); block 256 (8 waves, one 16-row q tile per wave).
// Register-prefetch of the next K/V chunk hides global latency behind WMMAs.
template <bool HAS_MASK>
__launch_bounds__(256)
__global__ void attn_kernel(const float* __restrict__ Q,
                            const float* __restrict__ Kg,
                            const float* __restrict__ Vg,
                            const float* __restrict__ maskb,
                            float* __restrict__ Og, int Tk) {
  __shared__ __align__(16) __bf16 Ks[64][72];       // [key][hd]
  __shared__ __align__(16) __bf16 Vt[64][72];       // [hd][key]
  __shared__ __align__(16) __bf16 Pl[8][16][72];    // per-wave P tile
  const int tid = threadIdx.x;
  const int w = tid >> 5;
  const int l = tid & 31;
  const int hi = l >> 4;
  const int ln16 = l & 15;
  const int b = blockIdx.x / HN;
  const int h = blockIdx.x % HN;
  const int qbase = blockIdx.y * 128 + w * 16;

  // Q fragments for this wave's 16 rows (hd = 64 -> 2 A-fragments)
  const float* qrow = Q + ((size_t)(b * LQN + qbase + ln16)) * DM + h * HDN;
  v16bf aq[2];
#pragma unroll
  for (int f = 0; f < 2; ++f)
#pragma unroll
    for (int e = 0; e < 16; ++e) {
      int hd = f * 32 + ((e >> 3) << 4) + (hi << 3) + (e & 7);
      aq[f][e] = (__bf16)qrow[hd];
    }

  float4 rk[4], rv[4];
  auto attn_loads = [&](int kb0) {
#pragma unroll
    for (int j = 0; j < 4; ++j) {        // 1024 float4 slots (64 keys x 16)
      int idx = tid + 256 * j;
      int key = idx >> 4;
      int c4 = (idx & 15) << 2;
      const size_t base = ((size_t)(b * Tk + kb0 + key)) * DM + h * HDN + c4;
      rk[j] = *reinterpret_cast<const float4*>(&Kg[base]);
      rv[j] = *reinterpret_cast<const float4*>(&Vg[base]);
    }
  };
  auto attn_store = [&]() {
#pragma unroll
    for (int j = 0; j < 4; ++j) {
      int idx = tid + 256 * j;
      int key = idx >> 4;
      int c4 = (idx & 15) << 2;
      Ks[key][c4 + 0] = (__bf16)rk[j].x;
      Ks[key][c4 + 1] = (__bf16)rk[j].y;
      Ks[key][c4 + 2] = (__bf16)rk[j].z;
      Ks[key][c4 + 3] = (__bf16)rk[j].w;
      Vt[c4 + 0][key] = (__bf16)rv[j].x;
      Vt[c4 + 1][key] = (__bf16)rv[j].y;
      Vt[c4 + 2][key] = (__bf16)rv[j].z;
      Vt[c4 + 3][key] = (__bf16)rv[j].w;
    }
  };

  v8f o[4];
  float m_r[8], l_r[8];
#pragma unroll
  for (int nt = 0; nt < 4; ++nt)
#pragma unroll
    for (int r = 0; r < 8; ++r) o[nt][r] = 0.0f;
#pragma unroll
  for (int r = 0; r < 8; ++r) { m_r[r] = -1.0e30f; l_r[r] = 0.0f; }

  const int nchunk = Tk >> 6;
  attn_loads(0);
  for (int c = 0; c < nchunk; ++c) {
    const int kb0 = c << 6;
    __syncthreads();
    attn_store();                        // consume regs of chunk c
    if (c + 1 < nchunk) attn_loads((c + 1) << 6);  // overlap with compute
    __syncthreads();

    // S = Q @ K^T for 4 key tiles of 16
    v8f s4[4];
#pragma unroll
    for (int nt = 0; nt < 4; ++nt) {
      v8f acc;
#pragma unroll
      for (int r = 0; r < 8; ++r) acc[r] = 0.0f;
      const __bf16* kr = &Ks[nt * 16 + ln16][0];
      v16bf bk0 = ld16(kr + 8 * hi, kr + 8 * hi + 16);
      v16bf bk1 = ld16(kr + 32 + 8 * hi, kr + 32 + 8 * hi + 16);
      acc = wmma_bf16(aq[0], bk0, acc);
      acc = wmma_bf16(aq[1], bk1, acc);
      s4[nt] = acc;
    }

    // online softmax (row reductions across 16-lane halves, wave32)
#pragma unroll
    for (int r = 0; r < 8; ++r) {
      const int qr = qbase + r + (hi << 3);
      float sv[4];
#pragma unroll
      for (int nt = 0; nt < 4; ++nt) {
        float v = s4[nt][r] * 0.125f;  // 1/sqrt(64)
        if (HAS_MASK)
          v += maskb[(size_t)qr * Tk + kb0 + nt * 16 + ln16];
        sv[nt] = v;
      }
      float mx = fmaxf(fmaxf(sv[0], sv[1]), fmaxf(sv[2], sv[3]));
#pragma unroll
      for (int off = 1; off < 16; off <<= 1)
        mx = fmaxf(mx, __shfl_xor(mx, off, 32));
      const float mn = fmaxf(m_r[r], mx);
      const float alpha = __expf(m_r[r] - mn);
      m_r[r] = mn;
      float ps = 0.0f;
#pragma unroll
      for (int nt = 0; nt < 4; ++nt) {
        float p = __expf(sv[nt] - mn);
        ps += p;
        Pl[w][r + (hi << 3)][nt * 16 + ln16] = (__bf16)p;
      }
#pragma unroll
      for (int off = 1; off < 16; off <<= 1) ps += __shfl_xor(ps, off, 32);
      l_r[r] = l_r[r] * alpha + ps;
#pragma unroll
      for (int nt = 0; nt < 4; ++nt) o[nt][r] *= alpha;
    }

    // O += P @ V  (2 key halves x 4 hd tiles)
#pragma unroll
    for (int f = 0; f < 2; ++f) {
      const __bf16* pr = &Pl[w][ln16][f * 32 + 8 * hi];
      v16bf ap = ld16(pr, pr + 16);
#pragma unroll
      for (int nt = 0; nt < 4; ++nt) {
        const __bf16* vr = &Vt[nt * 16 + ln16][f * 32 + 8 * hi];
        v16bf bv = ld16(vr, vr + 16);
        o[nt] = wmma_bf16(ap, bv, o[nt]);
      }
    }
  }

#pragma unroll
  for (int nt = 0; nt < 4; ++nt)
#pragma unroll
    for (int r = 0; r < 8; ++r) {
      const int qr = qbase + r + (hi << 3);
      const int col = h * HDN + nt * 16 + ln16;
      Og[((size_t)(b * LQN + qr)) * DM + col] = o[nt][r] / l_r[r];
    }
}

// ---------------- layernorm (one 128-thread block per row of D=512) ---------
__launch_bounds__(128)
__global__ void ln_kernel(const float* __restrict__ in, const float* __restrict__ g,
                          const float* __restrict__ b, float* __restrict__ out) {
  const int row = blockIdx.x;
  const int tid = threadIdx.x;
  const float* p = in + (size_t)row * DM;
  float4 v = *reinterpret_cast<const float4*>(&p[tid << 2]);
  float s = v.x + v.y + v.z + v.w;
  float q = v.x * v.x + v.y * v.y + v.z * v.z + v.w * v.w;
#pragma unroll
  for (int off = 1; off < 32; off <<= 1) {
    s += __shfl_xor(s, off, 32);
    q += __shfl_xor(q, off, 32);
  }
  __shared__ float ss[4], qq[4];
  if ((tid & 31) == 0) { ss[tid >> 5] = s; qq[tid >> 5] = q; }
  __syncthreads();
  s = ss[0] + ss[1] + ss[2] + ss[3];
  q = qq[0] + qq[1] + qq[2] + qq[3];
  const float mean = s * (1.0f / DM);
  const float var = q * (1.0f / DM) - mean * mean;
  const float rstd = rsqrtf(var + 1e-5f);
  float4 gv = *reinterpret_cast<const float4*>(&g[tid << 2]);
  float4 bv = *reinterpret_cast<const float4*>(&b[tid << 2]);
  float4 r;
  r.x = (v.x - mean) * rstd * gv.x + bv.x;
  r.y = (v.y - mean) * rstd * gv.y + bv.y;
  r.z = (v.z - mean) * rstd * gv.z + bv.z;
  r.w = (v.w - mean) * rstd * gv.w + bv.w;
  *reinterpret_cast<float4*>(&out[(size_t)row * DM + (tid << 2)]) = r;
}

// ---------------- mask: m[lq][t] = scale*(rel_bias[idx] + logg) -------------
__launch_bounds__(256)
__global__ void mask_kernel(const float* __restrict__ rel_row,
                            const float* __restrict__ scale_p,
                            float* __restrict__ maskb) {
  const int idx = blockIdx.x * 256 + threadIdx.x;
  const int lq = idx / TT;
  const int t = idx % TT;
  const float tau = (float)lq * (4095.0f / 255.0f);
  const float dt = (float)t - tau;
  float c = fminf(128.0f, fmaxf(-128.0f, dt));
  const int ri = (int)c + 128;  // truncation toward zero, like astype(int32)
  const float z = dt * (1.0f / 32.0f);
  const float logg = logf(expf(-0.5f * z * z) + 1e-6f);
  maskb[idx] = scale_p[0] * (rel_row[ri] + logg);
}

__launch_bounds__(256)
__global__ void bcast_kernel(const float* __restrict__ lat0, float* __restrict__ lat) {
  const int idx = blockIdx.x * 256 + threadIdx.x;
  lat[idx] = lat0[idx % (LQN * DM)];
}

__launch_bounds__(256)
__global__ void mean_kernel(const float* __restrict__ in, float* __restrict__ out) {
  const int idx = blockIdx.x * 256 + threadIdx.x;  // B*D total
  const int b = idx / DM;
  const int d = idx % DM;
  float s = 0.0f;
  for (int q = 0; q < LQN; ++q) s += in[((size_t)(b * LQN + q)) * DM + d];
  out[idx] = s * (1.0f / LQN);
}

// ---------------- host launcher ---------------------------------------------
static inline void gemm_launch(int act, int res, const float* A, const float* W,
                               const float* bias, const float* Res, float* C,
                               int M, int N, int K, hipStream_t s) {
  dim3 g(N / 64, M / 256);
  dim3 blk(256);
  if (act == 1)
    gemm_bias_kernel<1, 0><<<g, blk, 0, s>>>(A, W, bias, Res, C, M, N, K);
  else if (res)
    gemm_bias_kernel<0, 1><<<g, blk, 0, s>>>(A, W, bias, Res, C, M, N, K);
  else
    gemm_bias_kernel<0, 0><<<g, blk, 0, s>>>(A, W, bias, Res, C, M, N, K);
}

extern "C" void kernel_launch(void* const* d_in, const int* in_sizes, int n_in,
                              void* d_out, int out_size, void* d_ws, size_t ws_size,
                              hipStream_t stream) {
  (void)in_sizes; (void)n_in; (void)out_size; (void)ws_size;
  const float* tokens     = (const float*)d_in[0];
  const float* proj_w     = (const float*)d_in[1];
  const float* proj_b     = (const float*)d_in[2];
  const float* latents    = (const float*)d_in[3];
  const float* lnx_g      = (const float*)d_in[4];
  const float* lnx_b      = (const float*)d_in[5];
  const float* xw_qkv     = (const float*)d_in[6];
  const float* xb_qkv     = (const float*)d_in[7];
  const float* xw_o       = (const float*)d_in[8];
  const float* xb_o       = (const float*)d_in[9];
  const float* rel_bias   = (const float*)d_in[10];
  const float* mask_scale = (const float*)d_in[11];
  const float* mx_w1      = (const float*)d_in[12];
  const float* mx_b1      = (const float*)d_in[13];
  const float* mx_w2      = (const float*)d_in[14];
  const float* mx_b2      = (const float*)d_in[15];
  const float* lns_g      = (const float*)d_in[16];
  const float* lns_b      = (const float*)d_in[17];
  const float* sw_qkv     = (const float*)d_in[18];
  const float* sb_qkv     = (const float*)d_in[19];
  const float* sw_o       = (const float*)d_in[20];
  const float* sb_o       = (const float*)d_in[21];
  const float* ms_w1      = (const float*)d_in[22];
  const float* ms_b1      = (const float*)d_in[23];
  const float* ms_w2      = (const float*)d_in[24];
  const float* ms_b2      = (const float*)d_in[25];
  const float* lno_g      = (const float*)d_in[26];
  const float* lno_b      = (const float*)d_in[27];

  const size_t XS = (size_t)BB * TT * DM;    // 16,777,216
  const size_t LS = (size_t)BB * LQN * DM;   // 1,048,576
  const size_t HS = (size_t)BB * LQN * HIDN; // 4,194,304
  float* ws = (float*)d_ws;
  float* x     = ws;
  float* kb    = x + XS;
  float* vb    = kb + XS;
  float* qb    = vb + XS;
  float* lat   = qb + LS;
  float* latn  = lat + LS;
  float* ao    = latn + LS;
  float* hid   = ao + LS;
  float* maskb = hid + HS;

  const int MT = BB * TT;   // 32768 token rows
  const int ML = BB * LQN;  // 2048 latent rows

  // x = tokens @ proj_w^T + proj_b
  gemm_launch(0, 0, tokens, proj_w, proj_b, nullptr, x, MT, DM, DIN, stream);
  // lat = broadcast(latents)
  bcast_kernel<<<(BB * LQN * DM) / 256, 256, 0, stream>>>(latents, lat);

  for (int i = 0; i < NBK; ++i) {
    const float* wq = xw_qkv + (size_t)i * 3 * DM * DM;
    const float* wk = wq + (size_t)DM * DM;
    const float* wv = wk + (size_t)DM * DM;
    const float* bq = xb_qkv + (size_t)i * 3 * DM;
    const float* bk = bq + DM;
    const float* bv = bk + DM;

    // ---- cross attention ----
    ln_kernel<<<ML, 128, 0, stream>>>(lat, lnx_g + i * DM, lnx_b + i * DM, latn);
    gemm_launch(0, 0, latn, wq, bq, nullptr, qb, ML, DM, DM, stream);
    gemm_launch(0, 0, x, wk, bk, nullptr, kb, MT, DM, DM, stream);
    gemm_launch(0, 0, x, wv, bv, nullptr, vb, MT, DM, DM, stream);
    mask_kernel<<<(LQN * TT) / 256, 256, 0, stream>>>(rel_bias + i * 257,
                                                      mask_scale + i, maskb);
    attn_kernel<true><<<dim3(BB * HN, LQN / 128), 256, 0, stream>>>(
        qb, kb, vb, maskb, ao, TT);
    gemm_launch(0, 1, ao, xw_o + (size_t)i * DM * DM, xb_o + i * DM, lat, lat,
                ML, DM, DM, stream);

    // ---- cross MLP ----
    ln_kernel<<<ML, 128, 0, stream>>>(lat, lnx_g + i * DM, lnx_b + i * DM, latn);
    gemm_launch(1, 0, latn, mx_w1 + (size_t)i * HIDN * DM, mx_b1 + i * HIDN,
                nullptr, hid, ML, HIDN, DM, stream);
    gemm_launch(0, 1, hid, mx_w2 + (size_t)i * DM * HIDN, mx_b2 + i * DM, lat,
                lat, ML, DM, HIDN, stream);

    // ---- self attention ----
    const float* swq = sw_qkv + (size_t)i * 3 * DM * DM;
    const float* swk = swq + (size_t)DM * DM;
    const float* swv = swk + (size_t)DM * DM;
    const float* sbq = sb_qkv + (size_t)i * 3 * DM;
    const float* sbk = sbq + DM;
    const float* sbv = sbk + DM;
    ln_kernel<<<ML, 128, 0, stream>>>(lat, lns_g + i * DM, lns_b + i * DM, latn);
    gemm_launch(0, 0, latn, swq, sbq, nullptr, qb, ML, DM, DM, stream);
    gemm_launch(0, 0, latn, swk, sbk, nullptr, kb, ML, DM, DM, stream);
    gemm_launch(0, 0, latn, swv, sbv, nullptr, vb, ML, DM, DM, stream);
    attn_kernel<false><<<dim3(BB * HN, LQN / 128), 256, 0, stream>>>(
        qb, kb, vb, nullptr, ao, LQN);
    gemm_launch(0, 1, ao, sw_o + (size_t)i * DM * DM, sb_o + i * DM, lat, lat,
                ML, DM, DM, stream);

    // ---- self MLP ----
    ln_kernel<<<ML, 128, 0, stream>>>(lat, lns_g + i * DM, lns_b + i * DM, latn);
    gemm_launch(1, 0, latn, ms_w1 + (size_t)i * HIDN * DM, ms_b1 + i * HIDN,
                nullptr, hid, ML, HIDN, DM, stream);
    gemm_launch(0, 1, hid, ms_w2 + (size_t)i * DM * HIDN, ms_b2 + i * DM, lat,
                lat, ML, DM, HIDN, stream);
  }

  // final LN + mean over latents
  ln_kernel<<<ML, 128, 0, stream>>>(lat, lno_g, lno_b, latn);
  mean_kernel<<<(BB * DM) / 256, 256, 0, stream>>>(latn, (float*)d_out);
}
